// AMM_70403103916214
// MI455X (gfx1250) — matrix operand order repaired and checked by reading, not compiled, not run on hardware
//
#include <hip/hip_runtime.h>
#include <hip/hip_bf16.h>
#include <math.h>

#define HW     9216      // 96*96
#define CK     392       // 256 + 136
#define LDK    416       // K padded to 13*32
#define KT     13        // k-tiles of 32
#define NTILE  576       // 9216 / 16
#define NSPLIT 4         // streamed-dimension splits (occupancy)
#define TILES_PER_SPLIT (NTILE / NSPLIT)   // 144
#define LDSROW 424       // LDS panel row stride (bf16 elems); 848B -> conflict-free b128
#define CHUNKS 832       // 16 rows * 52 16-byte chunks per row (416*2/16)

typedef __attribute__((ext_vector_type(16))) __bf16 v16bf;
typedef __attribute__((ext_vector_type(8)))  __bf16 v8bf;
typedef __attribute__((ext_vector_type(8)))  float  v8f;

__device__ __forceinline__ unsigned short f2bf(float f) {
    unsigned int u = __float_as_uint(f);
    unsigned int r = (u + 0x7FFFu + ((u >> 16) & 1u)) >> 16;  // RNE
    return (unsigned short)r;
}

__device__ __forceinline__ v16bf cat8(v8bf lo, v8bf hi) {
    return __builtin_shufflevector(lo, hi, 0,1,2,3,4,5,6,7,8,9,10,11,12,13,14,15);
}

// Cooperatively stage a 16-row x 416-elem bf16 panel (row-major, stride LDK)
// into LDS (stride LDSROW). 256 threads, 16-byte chunks.
__device__ __forceinline__ void stage_panel(unsigned short* __restrict__ dst,
                                            const unsigned short* __restrict__ src,
                                            int row0, int tid) {
    #pragma unroll
    for (int t = 0; t < 4; ++t) {
        int c = tid + t * 256;
        if (c < CHUNKS) {
            int row = c / 52;
            int cc  = c - row * 52;
            uint4 v = *(const uint4*)(src + (size_t)(row0 + row) * LDK + cc * 8);
            *(uint4*)(dst + row * LDSROW + cc * 8) = v;
        }
    }
}

// ---------------- prep kernels ----------------

// S (9216 x 416 bf16 row-major): raw reshape S[i,k] = srcflat[i*392+k],
// srcflat = concat(0.01*feat_src, landmarks_src) in (c,hw) order.
__global__ void prep_S(const float* __restrict__ fs, const float* __restrict__ ls,
                       unsigned short* __restrict__ Sb) {
    int e = blockIdx.x * blockDim.x + threadIdx.x;
    if (e >= HW * LDK) return;
    int i = e / LDK, k = e - i * LDK;
    float v = 0.f;
    if (k < CK) {
        long q = (long)i * CK + k;
        v = (q < 256L * HW) ? 0.01f * fs[q] : ls[q - 256L * HW];
    }
    Sb[e] = f2bf(v);
}

// R^T (9216 x 416 bf16 row-major): Rt[j,k] = refflat[k*9216 + j].
__global__ void prep_RT(const float* __restrict__ fr, const float* __restrict__ lr,
                        unsigned short* __restrict__ Rt) {
    int e = blockIdx.x * blockDim.x + threadIdx.x;
    if (e >= HW * LDK) return;
    int j = e / LDK, k = e - j * LDK;
    float v = 0.f;
    if (k < 256)       v = 0.01f * fr[(size_t)k * HW + j];
    else if (k < CK)   v = lr[(size_t)(k - 256) * HW + j];
    Rt[e] = f2bf(v);
}

// beta[j], gama[j] (1x1 convs) pre-multiplied by mask equality.
__global__ void prep_vec(const float* __restrict__ fr,
                         const int* __restrict__ ms, const int* __restrict__ mr,
                         const float* __restrict__ w1, const float* __restrict__ b1,
                         const float* __restrict__ w2, const float* __restrict__ b2,
                         float* __restrict__ bm, float* __restrict__ gm) {
    int j = blockIdx.x * blockDim.x + threadIdx.x;
    if (j >= HW) return;
    float beta = b1[0], gama = b2[0];
    for (int c = 0; c < 256; ++c) {
        float f = fr[(size_t)c * HW + j];
        beta = fmaf(f, w1[c], beta);
        gama = fmaf(f, w2[c], gama);
    }
    float msk = (ms[j] == mr[j]) ? 1.f : 0.f;
    bm[j] = msk * beta;
    gm[j] = msk * gama;
}

// ---------------- pass 1: partial column softmax stats ----------------
// Grid (72, NSPLIT); block = 8 waves; wave owns ONE 16-col tile (B panel in
// 104 VGPRs); blockIdx.y selects which 144 row-tiles this block streams.
__global__ __launch_bounds__(256) void pass1_stats(
        const unsigned short* __restrict__ Sb,
        const unsigned short* __restrict__ Rt,
        float* __restrict__ mPart, float* __restrict__ sPart) {
    __shared__ unsigned short Apan[2][16 * LDSROW];
    const int tid  = threadIdx.x;
    const int lane = tid & 31;
    const int wave = tid >> 5;
    const int j0   = (blockIdx.x * 8 + wave) * 16;
    const int it0  = blockIdx.y * TILES_PER_SPLIT;
    const int half = lane >> 4;
    const int l15  = lane & 15;

    // B panel: column j0+l15, WMMA table-order K chunks; register-resident.
    v16bf breg[KT];
    {
        const unsigned short* brow = Rt + (size_t)(j0 + l15) * LDK + half * 8;
        #pragma unroll
        for (int kt = 0; kt < KT; ++kt)
            breg[kt] = cat8(*(const v8bf*)(brow + kt * 32), *(const v8bf*)(brow + kt * 32 + 16));
    }

    float mrun = -INFINITY, srun = 0.f;

    stage_panel(Apan[0], Sb, it0 * 16, tid);
    int p = 0;
    for (int it = 0; it < TILES_PER_SPLIT; ++it) {
        __syncthreads();
        if (it + 1 < TILES_PER_SPLIT)
            stage_panel(Apan[p ^ 1], Sb, (it0 + it + 1) * 16, tid);

        const unsigned short* lrow = Apan[p] + l15 * LDSROW + half * 8;
        v8f c = {};
        #pragma unroll
        for (int kt = 0; kt < KT; ++kt) {
            v8bf alo = *(const v8bf*)(lrow + kt * 32);
            v8bf ahi = *(const v8bf*)(lrow + kt * 32 + 16);
            c = __builtin_amdgcn_wmma_f32_16x16x32_bf16(false, cat8(alo, ahi),
                                                        false, breg[kt],
                                                        (short)0, c, false, false);
        }
        // online (max, sum-exp): lanes L and L+16 share a column.
        float tmax = c[0];
        #pragma unroll
        for (int r = 1; r < 8; ++r) tmax = fmaxf(tmax, c[r]);
        tmax = fmaxf(tmax, __shfl_xor(tmax, 16));
        float mnew = fmaxf(mrun, tmax);
        float ps = 0.f;
        #pragma unroll
        for (int r = 0; r < 8; ++r) ps += __expf(c[r] - mnew);
        ps += __shfl_xor(ps, 16);
        srun = srun * __expf(mrun - mnew) + ps;
        mrun = mnew;

        p ^= 1;
    }
    if (half == 0) {
        mPart[blockIdx.y * HW + j0 + l15] = mrun;
        sPart[blockIdx.y * HW + j0 + l15] = srun;
    }
}

// Merge NSPLIT partial (m,s) pairs, then fold mask*beta/gama / S.
__global__ void combine_stats(const float* __restrict__ mPart,
                              const float* __restrict__ sPart,
                              const float* __restrict__ bm, const float* __restrict__ gm,
                              float* __restrict__ mOut,
                              float* __restrict__ wb, float* __restrict__ wg) {
    int j = blockIdx.x * blockDim.x + threadIdx.x;
    if (j >= HW) return;
    float m = -INFINITY;
    #pragma unroll
    for (int s = 0; s < NSPLIT; ++s) m = fmaxf(m, mPart[s * HW + j]);
    float ssum = 0.f;
    #pragma unroll
    for (int s = 0; s < NSPLIT; ++s)
        ssum += sPart[s * HW + j] * __expf(mPart[s * HW + j] - m);
    float inv = 1.f / ssum;
    mOut[j] = m;
    wb[j] = bm[j] * inv;
    wg[j] = gm[j] * inv;
}

// ---------------- pass 2: partial fused softmax * mask * matvec ----------------
// Grid (72, NSPLIT); wave owns ONE 16-row tile (A panel in 104 VGPRs);
// blockIdx.y selects which 144 column-tiles stream through LDS.
__global__ __launch_bounds__(256) void pass2_acc(
        const unsigned short* __restrict__ Sb,
        const unsigned short* __restrict__ Rt,
        const float* __restrict__ mIn,
        const float* __restrict__ wbIn, const float* __restrict__ wgIn,
        float* __restrict__ bhatPart, float* __restrict__ ghatPart) {
    __shared__ unsigned short Bpan[2][16 * LDSROW];
    const int tid  = threadIdx.x;
    const int lane = tid & 31;
    const int wave = tid >> 5;
    const int i0   = (blockIdx.x * 8 + wave) * 16;
    const int jt0  = blockIdx.y * TILES_PER_SPLIT;
    const int half = lane >> 4;
    const int l15  = lane & 15;

    // A panel: row i0+l15; WMMA A layout chunks; register-resident.
    v16bf areg[KT];
    {
        const unsigned short* arow = Sb + (size_t)(i0 + l15) * LDK + half * 8;
        #pragma unroll
        for (int kt = 0; kt < KT; ++kt) {
            v8bf alo = *(const v8bf*)(arow + kt * 32);
            v8bf ahi = *(const v8bf*)(arow + kt * 32 + 16);
            areg[kt] = cat8(alo, ahi);
        }
    }

    float accb[8] = {}, accg[8] = {};

    stage_panel(Bpan[0], Rt, jt0 * 16, tid);
    int p = 0;
    for (int jt = 0; jt < TILES_PER_SPLIT; ++jt) {
        __syncthreads();
        if (jt + 1 < TILES_PER_SPLIT)
            stage_panel(Bpan[p ^ 1], Rt, (jt0 + jt + 1) * 16, tid);

        const int jj = (jt0 + jt) * 16 + l15;
        float mj  = mIn[jj];
        float wbj = wbIn[jj];
        float wgj = wgIn[jj];

        const unsigned short* lrow = Bpan[p] + l15 * LDSROW + half * 8;
        v8f c = {};
        #pragma unroll
        for (int kt = 0; kt < KT; ++kt) {
            v8bf blo = *(const v8bf*)(lrow + kt * 32);
            v8bf bhi = *(const v8bf*)(lrow + kt * 32 + 16);
            c = __builtin_amdgcn_wmma_f32_16x16x32_bf16(false, areg[kt], false,
                                                        cat8(blo, bhi),
                                                        (short)0, c, false, false);
        }
        #pragma unroll
        for (int r = 0; r < 8; ++r) {
            float e = __expf(c[r] - mj);
            accb[r] = fmaf(e, wbj, accb[r]);
            accg[r] = fmaf(e, wgj, accg[r]);
        }
        p ^= 1;
    }

    // reduce across the 16 column-lanes inside each half, then write rows.
    #pragma unroll
    for (int r = 0; r < 8; ++r) {
        float b = accb[r], g = accg[r];
        #pragma unroll
        for (int off = 1; off < 16; off <<= 1) {
            b += __shfl_xor(b, off);
            g += __shfl_xor(g, off);
        }
        if (l15 == 0) {
            int row = i0 + r + 8 * half;
            bhatPart[blockIdx.y * HW + row] = b;
            ghatPart[blockIdx.y * HW + row] = g;
        }
    }
}

__global__ void combine_hat(const float* __restrict__ bhatPart,
                            const float* __restrict__ ghatPart,
                            float* __restrict__ bhat, float* __restrict__ ghat) {
    int i = blockIdx.x * blockDim.x + threadIdx.x;
    if (i >= HW) return;
    float b = 0.f, g = 0.f;
    #pragma unroll
    for (int s = 0; s < NSPLIT; ++s) {
        b += bhatPart[s * HW + i];
        g += ghatPart[s * HW + i];
    }
    bhat[i] = b;
    ghat[i] = g;
}

__global__ void epilogue(const float* __restrict__ fs,
                         const float* __restrict__ betaHat,
                         const float* __restrict__ gamaHat,
                         float* __restrict__ out, int n) {
    int e = blockIdx.x * blockDim.x + threadIdx.x;
    if (e >= n) return;
    int p = e % HW;
    out[e] = fmaf(gamaHat[p], fs[e], betaHat[p]);
}

// ---------------- launcher ----------------
extern "C" void kernel_launch(void* const* d_in, const int* in_sizes, int n_in,
                              void* d_out, int out_size, void* d_ws, size_t ws_size,
                              hipStream_t stream) {
    (void)in_sizes; (void)n_in; (void)out_size; (void)ws_size;
    const float* feat_src = (const float*)d_in[0];
    const float* feat_ref = (const float*)d_in[1];
    const float* lm_src   = (const float*)d_in[2];
    const float* lm_ref   = (const float*)d_in[3];
    const int*   msrc     = (const int*)d_in[4];
    const int*   mref     = (const int*)d_in[5];
    const float* w1       = (const float*)d_in[6];
    const float* b1       = (const float*)d_in[7];
    const float* w2       = (const float*)d_in[8];
    const float* b2       = (const float*)d_in[9];
    float* out = (float*)d_out;

    char* ws = (char*)d_ws;
    unsigned short* Sb = (unsigned short*)ws;                          // 9216*416 bf16
    unsigned short* Rt = (unsigned short*)(ws + (size_t)HW * LDK * 2); // 9216*416 bf16
    float* fbase = (float*)(ws + (size_t)HW * LDK * 4);
    float* mPart = fbase;                       // NSPLIT * HW
    float* sPart = mPart + NSPLIT * HW;         // NSPLIT * HW
    float* bm    = sPart + NSPLIT * HW;         // HW
    float* gm    = bm + HW;                     // HW
    float* mstat = gm + HW;                     // HW
    float* wb    = mstat + HW;                  // HW
    float* wg    = wb + HW;                     // HW
    float* bhatP = wg + HW;                     // NSPLIT * HW
    float* ghatP = bhatP + NSPLIT * HW;         // NSPLIT * HW
    float* bhat  = ghatP + NSPLIT * HW;         // HW
    float* ghat  = bhat + HW;                   // HW

    const int nSR = HW * LDK;
    prep_S <<<(nSR + 255) / 256, 256, 0, stream>>>(feat_src, lm_src, Sb);
    prep_RT<<<(nSR + 255) / 256, 256, 0, stream>>>(feat_ref, lm_ref, Rt);
    prep_vec<<<(HW + 255) / 256, 256, 0, stream>>>(feat_ref, msrc, mref,
                                                   w1, b1, w2, b2, bm, gm);
    // 72 x NSPLIT blocks, 8 waves each: 2304 waves resident for latency hiding.
    pass1_stats<<<dim3(72, NSPLIT), 256, 0, stream>>>(Sb, Rt, mPart, sPart);
    combine_stats<<<(HW + 255) / 256, 256, 0, stream>>>(mPart, sPart, bm, gm,
                                                        mstat, wb, wg);
    pass2_acc<<<dim3(72, NSPLIT), 256, 0, stream>>>(Sb, Rt, mstat, wb, wg,
                                                    bhatP, ghatP);
    combine_hat<<<(HW + 255) / 256, 256, 0, stream>>>(bhatP, ghatP, bhat, ghat);
    const int nOut = 256 * HW;
    epilogue<<<(nOut + 255) / 256, 256, 0, stream>>>(feat_src, bhat, ghat, out, nOut);
}
